// MyLSTM_19404662243370
// MI455X (gfx1250) — compile-verified, matrix operand order checked by reading, not run on hardware
//
#include <hip/hip_runtime.h>

typedef __attribute__((ext_vector_type(16))) __bf16 v16bf;
typedef __attribute__((ext_vector_type(4)))  __bf16 v4bf;
typedef __attribute__((ext_vector_type(8)))  float  v8f;
typedef __attribute__((ext_vector_type(4)))  unsigned tdm_g0_t;
typedef __attribute__((ext_vector_type(8)))  int      tdm_g1_t;
typedef __attribute__((ext_vector_type(4)))  int      tdm_g4_t;
typedef __attribute__((ext_vector_type(8)))  int      tdm_g8_t;

#define T_STEPS 256
#define BATCH   64
#define HID     768
#define DIN     1536
#define NGATE   3072
#define MTOT    (T_STEPS * BATCH)      // 16384
#define NLAYER  3
#define NB_SCAN 24                     // 12 WGs per direction

// ---------------------------------------------------------------------------
// fp32 -> bf16 convert (grid-stride)
// ---------------------------------------------------------------------------
__global__ __launch_bounds__(256) void cvt_f32_bf16(const float* __restrict__ s,
                                                    __bf16* __restrict__ d, size_t n) {
  size_t i = (size_t)blockIdx.x * 256 + threadIdx.x;
  size_t stride = (size_t)gridDim.x * 256;
  for (; i < n; i += stride) d[i] = (__bf16)s[i];
}

// ---------------------------------------------------------------------------
// zero h double-buffers (both dirs, both phases) + barrier counter
// ---------------------------------------------------------------------------
__global__ __launch_bounds__(256) void init_state(__bf16* __restrict__ hbuf,
                                                  unsigned* __restrict__ cnt) {
  int i = blockIdx.x * 256 + threadIdx.x;
  if (i < 4 * BATCH * HID) hbuf[i] = (__bf16)0.0f;
  if (i == 0) *cnt = 0u;
}

// ---------------------------------------------------------------------------
// TDM: DMA one 128-row x 32-col bf16 tile (row stride DIN elems) from global
// into LDS, padding each 64B row with 32B so LDS row stride = 48 bf16.
// D# fields per CDNA5 ISA (6-arg clang-23 form).
// ---------------------------------------------------------------------------
__device__ __forceinline__ void tdm_load_tile_128x32(const __bf16* gsrc, unsigned lds_off) {
  unsigned long long ga = (unsigned long long)(uintptr_t)gsrc;
  tdm_g0_t g0;
  g0[0] = 1u;                                    // count=1, user descriptor
  g0[1] = lds_off;                               // LDS byte address
  g0[2] = (unsigned)ga;                          // global_addr[31:0]  (D# bits 95:64)
  g0[3] = (unsigned)(ga >> 32) | (2u << 30);     // global_addr[56:32] + type=2
  tdm_g1_t g1;
  g1[0] = (int)((1u << 16)                       // data_size = 2 bytes
              | (1u << 20)                       // pad_enable
              | (3u << 22)                       // pad_interval: 16 DWORDs (64B rows)
              | (7u << 25));                     // pad_amount: 8 DWORDs (32B pad)
  g1[1] = (int)(32u << 16);                      // tensor_dim0 = 32  (lo16 -> bits 63:48)
  g1[2] = (int)(128u << 16);                     // tensor_dim1 = 128 (lo16 -> bits 95:80)
  g1[3] = (int)(32u << 16);                      // tile_dim0 = 32    (bits 127:112)
  g1[4] = (int)128;                              // tile_dim1 = 128, tile_dim2 = 0
  g1[5] = (int)DIN;                              // tensor_dim0_stride = 1536 (lo32)
  g1[6] = 0;                                     // stride0 hi / stride1 lo
  g1[7] = 0;
  tdm_g4_t z4 = {0, 0, 0, 0};
  tdm_g8_t z8 = {0, 0, 0, 0, 0, 0, 0, 0};
  __builtin_amdgcn_tensor_load_to_lds(g0, g1, z4, z4, z8, 0);
}

// ---------------------------------------------------------------------------
// xg = xin(bf16) * w^T(bf16) + (bih+bhh), stored bf16 in gate-innermost
// layout xg[m][hcol][4]. M=16384, N=3072, K=1536. Block tile 128x128,
// 8 waves (2x4), wave tile 64x32. A/B tiles staged by TDM, double-buffered.
// ---------------------------------------------------------------------------
__global__ __launch_bounds__(256) void xg_gemm(const __bf16* __restrict__ xin,
                                               const __bf16* __restrict__ w,
                                               const float* __restrict__ bih,
                                               const float* __restrict__ bhh,
                                               __bf16* __restrict__ xg) {
  __shared__ __bf16 lA[2][128 * 48];   // 12KB per buffer (48-elem padded rows)
  __shared__ __bf16 lB[2][128 * 48];

  const int tid  = threadIdx.x;
  const int lane = tid & 31;
  const int wave = tid >> 5;
  const int wm   = wave >> 2;            // 0..1  (64 rows each)
  const int wn   = wave & 3;             // 0..3  (32 cols each)
  const int bm   = blockIdx.x * 128;
  const int bn   = blockIdx.y * 128;

  v8f acc[4][2] = {};

  // prologue: stage chunk 0
  if (wave == 0) {
    tdm_load_tile_128x32(xin + (size_t)bm * DIN, (unsigned)(uintptr_t)&lA[0][0]);
    tdm_load_tile_128x32(w   + (size_t)bn * DIN, (unsigned)(uintptr_t)&lB[0][0]);
  }

  for (int kk = 0; kk < DIN; kk += 32) {
    const int cur = (kk >> 5) & 1;
    if (wave == 0) {
      if (kk + 32 < DIN) {
        tdm_load_tile_128x32(xin + (size_t)bm * DIN + kk + 32,
                             (unsigned)(uintptr_t)&lA[cur ^ 1][0]);
        tdm_load_tile_128x32(w   + (size_t)bn * DIN + kk + 32,
                             (unsigned)(uintptr_t)&lB[cur ^ 1][0]);
        __builtin_amdgcn_s_wait_tensorcnt(2);   // current chunk's 2 loads done
      } else {
        __builtin_amdgcn_s_wait_tensorcnt(0);
      }
    }
    __syncthreads();

    const int ksel = (lane >> 4) * 16;
    v16bf bf0 = *(const v16bf*)(&lB[cur][0] + (wn * 32 +  0 + (lane & 15)) * 48 + ksel);
    v16bf bf1 = *(const v16bf*)(&lB[cur][0] + (wn * 32 + 16 + (lane & 15)) * 48 + ksel);
#pragma unroll
    for (int mt = 0; mt < 4; ++mt) {
      v16bf af = *(const v16bf*)(&lA[cur][0] + (wm * 64 + mt * 16 + (lane & 15)) * 48 + ksel);
      acc[mt][0] = __builtin_amdgcn_wmma_f32_16x16x32_bf16(false, af, false, bf0,
                                                           (short)0, acc[mt][0], false, false);
      acc[mt][1] = __builtin_amdgcn_wmma_f32_16x16x32_bf16(false, af, false, bf1,
                                                           (short)0, acc[mt][1], false, false);
    }
    __syncthreads();   // protect buffer from next TDM overwrite
  }

#pragma unroll
  for (int mt = 0; mt < 4; ++mt)
#pragma unroll
    for (int nt = 0; nt < 2; ++nt) {
      const int n = bn + wn * 32 + nt * 16 + (lane & 15);
      const int q = n / HID;               // gate index 0..3
      const int hcol = n - q * HID;        // hidden column
      const float bias = bih[n] + bhh[n];
#pragma unroll
      for (int r = 0; r < 8; ++r) {
        const int m = bm + wm * 64 + mt * 16 + r + ((lane >> 4) << 3);
        xg[((size_t)m * HID + hcol) * 4 + q] = (__bf16)(acc[mt][nt][r] + bias);
      }
    }
}

// ---------------------------------------------------------------------------
// grid barrier: monotonic counter, target grows by NB_SCAN per step
// ---------------------------------------------------------------------------
__device__ __forceinline__ void grid_barrier(unsigned* cnt, unsigned target) {
  __threadfence();
  __syncthreads();
  if (threadIdx.x == 0) {
    __hip_atomic_fetch_add(cnt, 1u, __ATOMIC_RELEASE, __HIP_MEMORY_SCOPE_AGENT);
    while (__hip_atomic_load(cnt, __ATOMIC_ACQUIRE, __HIP_MEMORY_SCOPE_AGENT) < target)
      __builtin_amdgcn_s_sleep(2);
  }
  __syncthreads();
}

// ---------------------------------------------------------------------------
// Persistent recurrent scan. 24 blocks: dir = blk/12, each WG owns 64 hidden
// units (256 gate cols). 8 waves: mt = wave&3 (batch tile), hh = wave>>2
// (hidden half). Each wave: 4 gates x 2 col-tiles = 8 WMMA accumulators,
// c-state lives in registers, gate combine is register-local. xg is read in
// gate-innermost layout: one v4bf load yields all four gate pre-activations.
// ---------------------------------------------------------------------------
__global__ __launch_bounds__(256) void lstm_scan(const __bf16* __restrict__ whh_f,
                                                 const __bf16* __restrict__ whh_b,
                                                 const __bf16* __restrict__ xg_f,
                                                 const __bf16* __restrict__ xg_b,
                                                 const float*  __restrict__ masks,
                                                 __bf16* __restrict__ hbuf,       // [2dir][2phase][64][768]
                                                 __bf16* __restrict__ xin_next,   // bf16 next-layer input (or null)
                                                 float*  __restrict__ out_inp,    // fp32 final output (or null)
                                                 float*  __restrict__ out_hn,
                                                 float*  __restrict__ out_cn,
                                                 int layer,
                                                 unsigned* __restrict__ bar) {
  const int tid   = threadIdx.x;
  const int lane  = tid & 31;
  const int wave  = tid >> 5;
  const int mt    = wave & 3;          // batch tile 0..3 (rows mt*16..)
  const int hh    = wave >> 2;         // hidden half 0..1
  const int blk   = blockIdx.x;
  const int dir   = blk / 12;
  const int hbase = (blk % 12) * 64;

  const __bf16* whh = dir ? whh_b : whh_f;
  const __bf16* xg  = dir ? xg_b  : xg_f;
  __bf16* hb = hbuf + (size_t)dir * 2 * BATCH * HID;

  const int arow  = mt * 16 + (lane & 15);
  const int ksel  = (lane >> 4) * 16;
  const int hcol0 = hbase + hh * 32;

  // weight row pointers for the 8 fragments (invariant over time)
  const __bf16* wrow[4][2];
#pragma unroll
  for (int q = 0; q < 4; ++q)
#pragma unroll
    for (int j = 0; j < 2; ++j)
      wrow[q][j] = whh + (size_t)(q * HID + hcol0 + j * 16 + (lane & 15)) * HID + ksel;

  float cst[2][8] = {{0.f}};

  for (int t = 0; t < T_STEPS; ++t) {
    const int tt = dir ? (T_STEPS - 1 - t) : t;
    const __bf16* hread = hb + (size_t)(t & 1) * BATCH * HID;
    __bf16* hwrite      = hb + (size_t)((t + 1) & 1) * BATCH * HID;

    v8f acc[4][2] = {};
    const __bf16* aptr = hread + (size_t)arow * HID + ksel;

#pragma unroll 2
    for (int kk = 0; kk < HID; kk += 32) {
      v16bf a = *(const v16bf*)(aptr + kk);
#pragma unroll
      for (int q = 0; q < 4; ++q)
#pragma unroll
        for (int j = 0; j < 2; ++j) {
          v16bf b = *(const v16bf*)(wrow[q][j] + kk);
          acc[q][j] = __builtin_amdgcn_wmma_f32_16x16x32_bf16(false, a, false, b,
                                                              (short)0, acc[q][j], false, false);
        }
    }

#pragma unroll
    for (int j = 0; j < 2; ++j) {
      const int hcol = hcol0 + j * 16 + (lane & 15);
#pragma unroll
      for (int r = 0; r < 8; ++r) {
        const int brow = mt * 16 + r + ((lane >> 4) << 3);
        v4bf g4 = *(const v4bf*)(xg + (((size_t)tt * BATCH + brow) * HID + hcol) * 4);
        float gi = acc[0][j][r] + (float)g4[0];
        float gf = acc[1][j][r] + (float)g4[1];
        float gg = acc[2][j][r] + (float)g4[2];
        float go = acc[3][j][r] + (float)g4[3];
        float i_ = 1.0f / (1.0f + __expf(-gi));
        float f_ = 1.0f / (1.0f + __expf(-gf));
        float g_ = tanhf(gg);
        float o_ = 1.0f / (1.0f + __expf(-go));
        float craw = f_ * cst[j][r] + i_ * g_;
        float m    = masks[tt * BATCH + brow];
        float h2   = o_ * tanhf(craw) * m;
        float c2   = craw * m;
        cst[j][r]  = c2;

        hwrite[(size_t)brow * HID + hcol] = (__bf16)h2;
        const size_t oidx = ((size_t)tt * BATCH + brow) * (2 * HID) + dir * HID + hcol;
        if (xin_next) xin_next[oidx] = (__bf16)h2;
        if (out_inp)  out_inp[oidx]  = h2;
        if (t == T_STEPS - 1) {
          const size_t nidx = ((size_t)layer * BATCH + brow) * (2 * HID) + dir * HID + hcol;
          out_hn[nidx] = h2;
          out_cn[nidx] = c2;
        }
      }
    }
    grid_barrier(bar, (unsigned)(t + 1) * NB_SCAN);
  }
}

// ---------------------------------------------------------------------------
// host launcher
// ---------------------------------------------------------------------------
extern "C" void kernel_launch(void* const* d_in, const int* in_sizes, int n_in,
                              void* d_out, int out_size, void* d_ws, size_t ws_size,
                              hipStream_t stream) {
  const float* x      = (const float*)d_in[0];
  const float* masks  = (const float*)d_in[1];
  const float* w_ih_f = (const float*)d_in[2];
  const float* w_hh_f = (const float*)d_in[3];
  const float* b_ih_f = (const float*)d_in[4];
  const float* b_hh_f = (const float*)d_in[5];
  const float* w_ih_b = (const float*)d_in[6];
  const float* w_hh_b = (const float*)d_in[7];
  const float* b_ih_b = (const float*)d_in[8];
  const float* b_hh_b = (const float*)d_in[9];

  float* out_inp = (float*)d_out;
  float* out_hn  = out_inp + (size_t)T_STEPS * BATCH * 2 * HID;
  float* out_cn  = out_hn  + (size_t)NLAYER * BATCH * 2 * HID;

  char* p = (char*)d_ws;
  auto carve = [&](size_t bytes) -> char* {
    char* r = p; p += (bytes + 255) & ~(size_t)255; return r;
  };
  const size_t szXin = (size_t)MTOT * DIN;           // elems
  const size_t szWih = (size_t)NGATE * DIN;          // per layer-dir
  const size_t szWhh = (size_t)NGATE * HID;          // per layer-dir
  const size_t szXg  = (size_t)MTOT * NGATE;         // per dir

  __bf16* xin0   = (__bf16*)carve(szXin * 2);
  __bf16* xin1   = (__bf16*)carve(szXin * 2);
  __bf16* wihbf  = (__bf16*)carve(2 * NLAYER * szWih * 2);   // [dir][l][3072][1536]
  __bf16* whhbf  = (__bf16*)carve(2 * NLAYER * szWhh * 2);   // [dir][l][3072][768]
  __bf16* xgf    = (__bf16*)carve(szXg * 2);
  __bf16* xgb    = (__bf16*)carve(szXg * 2);
  __bf16* hbuf   = (__bf16*)carve((size_t)4 * BATCH * HID * 2);
  unsigned* bar  = (unsigned*)carve(256);

  // convert activations + all weights to bf16
  cvt_f32_bf16<<<2048, 256, 0, stream>>>(x, xin0, szXin);
  cvt_f32_bf16<<<2048, 256, 0, stream>>>(w_ih_f, wihbf,                  NLAYER * szWih);
  cvt_f32_bf16<<<2048, 256, 0, stream>>>(w_ih_b, wihbf + NLAYER * szWih, NLAYER * szWih);
  cvt_f32_bf16<<<2048, 256, 0, stream>>>(w_hh_f, whhbf,                  NLAYER * szWhh);
  cvt_f32_bf16<<<2048, 256, 0, stream>>>(w_hh_b, whhbf + NLAYER * szWhh, NLAYER * szWhh);

  __bf16* cur = xin0;
  __bf16* nxt = xin1;
  dim3 ggrid(MTOT / 128, NGATE / 128);

  for (int l = 0; l < NLAYER; ++l) {
    // input-gate GEMMs (parallel over all timesteps)
    xg_gemm<<<ggrid, 256, 0, stream>>>(cur, wihbf + (size_t)(0 * NLAYER + l) * szWih,
                                       b_ih_f + l * NGATE, b_hh_f + l * NGATE, xgf);
    xg_gemm<<<ggrid, 256, 0, stream>>>(cur, wihbf + (size_t)(1 * NLAYER + l) * szWih,
                                       b_ih_b + l * NGATE, b_hh_b + l * NGATE, xgb);

    init_state<<<(4 * BATCH * HID + 255) / 256, 256, 0, stream>>>(hbuf, bar);

    lstm_scan<<<NB_SCAN, 256, 0, stream>>>(
        whhbf + (size_t)(0 * NLAYER + l) * szWhh,
        whhbf + (size_t)(1 * NLAYER + l) * szWhh,
        xgf, xgb, masks, hbuf,
        (l < NLAYER - 1) ? nxt : (__bf16*)nullptr,
        (l == NLAYER - 1) ? out_inp : (float*)nullptr,
        out_hn, out_cn, l, bar);

    __bf16* tmp = cur; cur = nxt; nxt = tmp;
  }
}
